// StrippingLayer_75608604278817
// MI455X (gfx1250) — matrix operand order repaired and checked by reading, not compiled, not run on hardware
//
#include <hip/hip_runtime.h>

// Problem constants: x (16,1024,2048) f32, codebooks (8,1024,1024) f32
#define NB    16
#define DIMD  1024
#define TT    2048
#define BINS  1024
#define NQ    8

typedef __attribute__((ext_vector_type(4)))  float  f4;
typedef __attribute__((ext_vector_type(8)))  float  v8f;
typedef __attribute__((ext_vector_type(16))) __bf16 v16bf;
typedef __attribute__((ext_vector_type(8)))  __bf16 bf8v;
typedef __attribute__((ext_vector_type(2)))  __bf16 bf2v;

__device__ __forceinline__ void splitf(float f, __bf16& h, __bf16& l) {
  __bf16 hh = (__bf16)f;
  h = hh;
  l = (__bf16)(f - (float)hh);
}

// ============ prep: codebook bf16 hi/lo split + per-bin squared norms =========
__global__ __launch_bounds__(256) void rvq_prep_kernel(const float* __restrict__ cb,
                                                       float* __restrict__ norms,
                                                       __bf16* __restrict__ cbh,
                                                       __bf16* __restrict__ cbl,
                                                       int do_split) {
  __shared__ float sred[256];
  const int row = blockIdx.x;                    // 0 .. NQ*BINS-1
  const float* p = cb + (size_t)row * DIMD;
  const int tid = threadIdx.x;
  float s = 0.f;
  for (int i = tid; i < DIMD; i += 256) {
    float v = p[i];
    s = fmaf(v, v, s);
    if (do_split) {
      __bf16 h, l; splitf(v, h, l);
      cbh[(size_t)row * DIMD + i] = h;
      cbl[(size_t)row * DIMD + i] = l;
    }
  }
  sred[tid] = s; __syncthreads();
  for (int off = 128; off >= 1; off >>= 1) {
    if (tid < off) sred[tid] += sred[tid + off];
    __syncthreads();
  }
  if (tid == 0) norms[row] = sred[0];
}

// ============ FAST main: 32-row tile, pre-split bf16 A in LDS, bf16 B global ==
#define ROWS2 32
#define RP    1028          // fp32 LDS row stride (bank-conflict-free b128)
#define BP    1032          // bf16 LDS row stride

__global__ __launch_bounds__(256) void rvq_main32_kernel(const float* __restrict__ x,
                                                         const float* __restrict__ cb,
                                                         const float* __restrict__ cbnorm,
                                                         const __bf16* __restrict__ cbh,
                                                         const __bf16* __restrict__ cbl,
                                                         float* __restrict__ outq,
                                                         float* __restrict__ outr,
                                                         float* __restrict__ outc,
                                                         float* __restrict__ partial) {
  __shared__ float  rf[ROWS2 * RP];              // fp32 residual tile
  __shared__ __bf16 ahi[ROWS2 * BP];             // bf16 hi of residual
  __shared__ __bf16 alo[ROWS2 * BP];             // bf16 lo of residual
  __shared__ float  redD[8][32];
  __shared__ int    redI[8][32];
  __shared__ int    codeSm[32];
  __shared__ float  wsum[8];

  const int tid  = threadIdx.x;
  const int lane = tid & 31;
  const int wave = tid >> 5;
  const int nlo  = lane & 15;
  const int h    = lane >> 4;
  const int bid  = blockIdx.x;                   // 0 .. 1023
  const int b    = bid >> 6;                     // T/32 = 64 tiles per batch
  const int t0   = (bid & 63) << 5;

  const float* xb = x + (size_t)b * DIMD * TT + t0;

  // ---- transpose-load x[b, :, t0:t0+32] -> rf[t][d]
  {
    const int j  = tid & 31;
    const int dh = tid >> 5;                     // 0..7
#pragma unroll 4
    for (int it = 0; it < 128; ++it) {
      int d = it * 8 + dh;
      rf[j * RP + d] = xb[(size_t)d * TT + j];
    }
  }
  __syncthreads();

  float lsum = 0.f;

  for (int s = 0; s < NQ; ++s) {
    const float* cbs = cb + (size_t)s * BINS * DIMD;

    // ---- split residual to bf16 hi/lo once per stage (reused by all N-tiles)
    {
      const int dbase = tid * 4;
#pragma unroll 1
      for (int m = 0; m < ROWS2; ++m) {
        float f0 = rf[m * RP + dbase + 0], f1 = rf[m * RP + dbase + 1];
        float f2 = rf[m * RP + dbase + 2], f3 = rf[m * RP + dbase + 3];
        __bf16 h0, l0, h1, l1, h2, l2, h3, l3;
        splitf(f0, h0, l0); splitf(f1, h1, l1);
        splitf(f2, h2, l2); splitf(f3, h3, l3);
        bf2v hv0, lv0, hv1, lv1;
        hv0[0] = h0; hv0[1] = h1; hv1[0] = h2; hv1[1] = h3;
        lv0[0] = l0; lv0[1] = l1; lv1[0] = l2; lv1[1] = l3;
        *(bf2v*)(&ahi[m * BP + dbase])     = hv0;
        *(bf2v*)(&ahi[m * BP + dbase + 2]) = hv1;
        *(bf2v*)(&alo[m * BP + dbase])     = lv0;
        *(bf2v*)(&alo[m * BP + dbase + 2]) = lv1;
      }
    }
    __syncthreads();

    float minv[2][8]; int mini[2][8];
#pragma unroll
    for (int mt = 0; mt < 2; ++mt)
#pragma unroll
      for (int v = 0; v < 8; ++v) { minv[mt][v] = 3.4e38f; mini[mt][v] = 0; }

    // each wave: 128 bins = 8 N-tiles; each B fragment feeds both m-tiles
    for (int nt = 0; nt < 8; ++nt) {
      const int n0 = wave * 128 + nt * 16;
      v8f acc0 = {}, acc1 = {};
      const __bf16* browh = cbh + (size_t)(s * BINS + n0 + nlo) * DIMD + h * 16;
      const __bf16* browl = cbl + (size_t)(s * BINS + n0 + nlo) * DIMD + h * 16;
      const __bf16* ar0h = &ahi[nlo * BP + h * 16];
      const __bf16* ar0l = &alo[nlo * BP + h * 16];
      const __bf16* ar1h = &ahi[(16 + nlo) * BP + h * 16];
      const __bf16* ar1l = &alo[(16 + nlo) * BP + h * 16];

#pragma unroll 2
      for (int ks = 0; ks < 32; ++ks) {
        v16bf bh = *(const v16bf*)(browh + ks * 32);
        v16bf bl = *(const v16bf*)(browl + ks * 32);
        v16bf a0h = __builtin_shufflevector(*(const bf8v*)(ar0h + ks * 32),
                                            *(const bf8v*)(ar0h + ks * 32 + 8),
                                            0,1,2,3,4,5,6,7,8,9,10,11,12,13,14,15);
        v16bf a0l = __builtin_shufflevector(*(const bf8v*)(ar0l + ks * 32),
                                            *(const bf8v*)(ar0l + ks * 32 + 8),
                                            0,1,2,3,4,5,6,7,8,9,10,11,12,13,14,15);
        v16bf a1h = __builtin_shufflevector(*(const bf8v*)(ar1h + ks * 32),
                                            *(const bf8v*)(ar1h + ks * 32 + 8),
                                            0,1,2,3,4,5,6,7,8,9,10,11,12,13,14,15);
        v16bf a1l = __builtin_shufflevector(*(const bf8v*)(ar1l + ks * 32),
                                            *(const bf8v*)(ar1l + ks * 32 + 8),
                                            0,1,2,3,4,5,6,7,8,9,10,11,12,13,14,15);
        acc0 = __builtin_amdgcn_wmma_f32_16x16x32_bf16(false, a0h, false, bh, (short)0, acc0, false, false);
        acc0 = __builtin_amdgcn_wmma_f32_16x16x32_bf16(false, a0h, false, bl, (short)0, acc0, false, false);
        acc0 = __builtin_amdgcn_wmma_f32_16x16x32_bf16(false, a0l, false, bh, (short)0, acc0, false, false);
        acc1 = __builtin_amdgcn_wmma_f32_16x16x32_bf16(false, a1h, false, bh, (short)0, acc1, false, false);
        acc1 = __builtin_amdgcn_wmma_f32_16x16x32_bf16(false, a1h, false, bl, (short)0, acc1, false, false);
        acc1 = __builtin_amdgcn_wmma_f32_16x16x32_bf16(false, a1l, false, bh, (short)0, acc1, false, false);
      }

      const float cn  = cbnorm[s * BINS + n0 + nlo];
      const int   idx = n0 + nlo;
#pragma unroll
      for (int v = 0; v < 8; ++v) {
        float d0 = fmaf(-2.f, acc0[v], cn);
        if (d0 < minv[0][v]) { minv[0][v] = d0; mini[0][v] = idx; }
        float d1 = fmaf(-2.f, acc1[v], cn);
        if (d1 < minv[1][v]) { minv[1][v] = d1; mini[1][v] = idx; }
      }
    }

    // cross-lane argmin within 16-lane halves (row = mt*16 + v + 8*h)
#pragma unroll
    for (int mt = 0; mt < 2; ++mt)
#pragma unroll
      for (int v = 0; v < 8; ++v) {
        float bd = minv[mt][v]; int bi = mini[mt][v];
#pragma unroll
        for (int m = 8; m >= 1; m >>= 1) {
          float od = __shfl_xor(bd, m, 16);
          int   oi = __shfl_xor(bi, m, 16);
          if (od < bd || (od == bd && oi < bi)) { bd = od; bi = oi; }
        }
        if (nlo == 0) { int row = mt * 16 + v + 8 * h; redD[wave][row] = bd; redI[wave][row] = bi; }
      }
    __syncthreads();

    if (tid < 32) {
      float bd = redD[0][tid]; int bi = redI[0][tid];
#pragma unroll
      for (int w = 1; w < 8; ++w) {
        float od = redD[w][tid]; int oi = redI[w][tid];
        if (od < bd || (od == bd && oi < bi)) { bd = od; bi = oi; }
      }
      codeSm[tid] = bi;
      outc[(size_t)s * NB * TT + (size_t)b * TT + t0 + tid] = (float)bi;
    }
    __syncthreads();

    // residual update r -= cb[code]; loss_i = mean(r_new^2)
    {
      const int dbase = tid * 4;
#pragma unroll 1
      for (int m = 0; m < ROWS2; ++m) {
        const float* q  = cbs + (size_t)codeSm[m] * DIMD + dbase;
        float*       rr = &rf[m * RP + dbase];
#pragma unroll
        for (int u = 0; u < 4; ++u) {
          float rv = rr[u] - q[u];
          rr[u] = rv;
          lsum = fmaf(rv, rv, lsum);
        }
      }
    }
    __syncthreads();
  }

  // loss partial: fixed-pattern wave butterfly + fixed-order 8-way sum
#pragma unroll
  for (int m = 16; m >= 1; m >>= 1) lsum += __shfl_xor(lsum, m, 32);
  if (lane == 0) wsum[wave] = lsum;
  __syncthreads();
  if (tid == 0) {
    float t = 0.f;
#pragma unroll
    for (int w = 0; w < 8; ++w) t += wsum[w];
    partial[bid] = t;
  }

  // outputs: residual_out = r_final ; quantized = x - r_final (both (B,D,T))
  {
    const int j  = tid & 31;
    const int dh = tid >> 5;
#pragma unroll 4
    for (int it = 0; it < 128; ++it) {
      int d = it * 8 + dh;
      float rv = rf[j * RP + d];
      size_t off = (size_t)b * DIMD * TT + (size_t)d * TT + t0 + j;
      outr[off] = rv;
      outq[off] = xb[(size_t)d * TT + j] - rv;
    }
  }
}

// ============ FALLBACK main (small ws): 16-row tile, in-loop split ============
#define ROWS 16
__device__ __forceinline__ void split16(const f4 a, const f4 b, const f4 c, const f4 d,
                                        v16bf& hi, v16bf& lo) {
  float tmp[16];
#pragma unroll
  for (int i = 0; i < 4; ++i) { tmp[i] = a[i]; tmp[4+i] = b[i]; tmp[8+i] = c[i]; tmp[12+i] = d[i]; }
#pragma unroll
  for (int i = 0; i < 16; ++i) {
    __bf16 h, l; splitf(tmp[i], h, l);
    hi[i] = h; lo[i] = l;
  }
}

__global__ __launch_bounds__(256) void rvq_main16_kernel(const float* __restrict__ x,
                                                         const float* __restrict__ cb,
                                                         const float* __restrict__ cbnorm,
                                                         float* __restrict__ outq,
                                                         float* __restrict__ outr,
                                                         float* __restrict__ outc,
                                                         float* __restrict__ partial) {
  __shared__ float rf[ROWS * RP];
  __shared__ float redD[8][16];
  __shared__ int   redI[8][16];
  __shared__ int   codeSm[16];
  __shared__ float wsum[8];

  const int tid  = threadIdx.x;
  const int lane = tid & 31;
  const int wave = tid >> 5;
  const int nlo  = lane & 15;
  const int h    = lane >> 4;
  const int bid  = blockIdx.x;
  const int b    = bid >> 7;
  const int t0   = (bid & 127) << 4;

  const float* xb = x + (size_t)b * DIMD * TT + t0;

  {
    const int j  = tid & 15;
    const int dh = tid >> 4;
#pragma unroll 4
    for (int it = 0; it < 64; ++it) {
      int d = it * 16 + dh;
      rf[j * RP + d] = xb[(size_t)d * TT + j];
    }
  }
  __syncthreads();

  float lsum = 0.f;

  for (int s = 0; s < NQ; ++s) {
    const float* cbs = cb + (size_t)s * BINS * DIMD;
    float minv[8]; int mini[8];
#pragma unroll
    for (int v = 0; v < 8; ++v) { minv[v] = 3.4e38f; mini[v] = 0; }

    for (int nt = 0; nt < 8; ++nt) {
      const int n0 = wave * 128 + nt * 16;
      v8f acc = {};
      const float* brow = cbs + (size_t)(n0 + nlo) * DIMD + h * 16;
      const float* arow = &rf[nlo * RP + h * 16];
#pragma unroll 2
      for (int ks = 0; ks < 32; ++ks) {
        const float* ap = arow + ks * 32;
        v16bf ahiV, aloV, bhiV, bloV;
        split16(*(const f4*)(ap), *(const f4*)(ap + 4),
                *(const f4*)(ap + 8), *(const f4*)(ap + 12), ahiV, aloV);
        const float* bp = brow + ks * 32;
        split16(*(const f4*)(bp), *(const f4*)(bp + 4),
                *(const f4*)(bp + 8), *(const f4*)(bp + 12), bhiV, bloV);
        acc = __builtin_amdgcn_wmma_f32_16x16x32_bf16(false, ahiV, false, bhiV, (short)0, acc, false, false);
        acc = __builtin_amdgcn_wmma_f32_16x16x32_bf16(false, ahiV, false, bloV, (short)0, acc, false, false);
        acc = __builtin_amdgcn_wmma_f32_16x16x32_bf16(false, aloV, false, bhiV, (short)0, acc, false, false);
      }
      const float cn  = cbnorm[s * BINS + n0 + nlo];
      const int   idx = n0 + nlo;
#pragma unroll
      for (int v = 0; v < 8; ++v) {
        float dst = fmaf(-2.f, acc[v], cn);
        if (dst < minv[v]) { minv[v] = dst; mini[v] = idx; }
      }
    }

#pragma unroll
    for (int v = 0; v < 8; ++v) {
      float bd = minv[v]; int bi = mini[v];
#pragma unroll
      for (int m = 8; m >= 1; m >>= 1) {
        float od = __shfl_xor(bd, m, 16);
        int   oi = __shfl_xor(bi, m, 16);
        if (od < bd || (od == bd && oi < bi)) { bd = od; bi = oi; }
      }
      if (nlo == 0) { int row = v + 8 * h; redD[wave][row] = bd; redI[wave][row] = bi; }
    }
    __syncthreads();

    if (tid < 16) {
      float bd = redD[0][tid]; int bi = redI[0][tid];
#pragma unroll
      for (int w = 1; w < 8; ++w) {
        float od = redD[w][tid]; int oi = redI[w][tid];
        if (od < bd || (od == bd && oi < bi)) { bd = od; bi = oi; }
      }
      codeSm[tid] = bi;
      outc[(size_t)s * NB * TT + (size_t)b * TT + t0 + tid] = (float)bi;
    }
    __syncthreads();

    {
      const int dbase = tid * 4;
#pragma unroll 1
      for (int m = 0; m < 16; ++m) {
        const float* q  = cbs + (size_t)codeSm[m] * DIMD + dbase;
        float*       rr = &rf[m * RP + dbase];
#pragma unroll
        for (int u = 0; u < 4; ++u) {
          float rv = rr[u] - q[u];
          rr[u] = rv;
          lsum = fmaf(rv, rv, lsum);
        }
      }
    }
    __syncthreads();
  }

#pragma unroll
  for (int m = 16; m >= 1; m >>= 1) lsum += __shfl_xor(lsum, m, 32);
  if (lane == 0) wsum[wave] = lsum;
  __syncthreads();
  if (tid == 0) {
    float t = 0.f;
#pragma unroll
    for (int w = 0; w < 8; ++w) t += wsum[w];
    partial[bid] = t;
  }

  {
    const int j  = tid & 15;
    const int dh = tid >> 4;
#pragma unroll 4
    for (int it = 0; it < 64; ++it) {
      int d = it * 16 + dh;
      float rv = rf[j * RP + d];
      size_t off = (size_t)b * DIMD * TT + (size_t)d * TT + t0 + j;
      outr[off] = rv;
      outq[off] = xb[(size_t)d * TT + j] - rv;
    }
  }
}

// ============ final: deterministic reduction of partials -> commit_loss ======
__global__ __launch_bounds__(256) void rvq_loss_kernel(const float* __restrict__ partial,
                                                       int count,
                                                       float* __restrict__ out_loss) {
  __shared__ float sred[256];
  const int tid = threadIdx.x;
  float s = 0.f;
  for (int i = tid; i < count; i += 256) s += partial[i];
  sred[tid] = s; __syncthreads();
  for (int off = 128; off >= 1; off >>= 1) {
    if (tid < off) sred[tid] += sred[tid + off];
    __syncthreads();
  }
  if (tid == 0) out_loss[0] = sred[0] * (1.0f / (8.0f * 16.0f * 2048.0f * 1024.0f));
}

extern "C" void kernel_launch(void* const* d_in, const int* in_sizes, int n_in,
                              void* d_out, int out_size, void* d_ws, size_t ws_size,
                              hipStream_t stream) {
  const float* x  = (const float*)d_in[0];   // (16,1024,2048)
  const float* cb = (const float*)d_in[1];   // (8,1024,1024)
  float* out = (float*)d_out;
  // d_out: quantized (33554432) | residual (33554432) | loss (1) | codes (262144)
  float* outq    = out;
  float* outr    = out + 33554432ull;
  float* outloss = out + 67108864ull;
  float* outc    = out + 67108865ull;
  // d_ws: norms (8192 f32) | partials (2048 f32) | [cb_hi | cb_lo bf16, 32MB]
  float*  norms   = (float*)d_ws;
  float*  partial = norms + 8192;
  __bf16* cbh     = (__bf16*)((char*)d_ws + 40960);
  __bf16* cbl     = cbh + (size_t)NQ * BINS * DIMD;

  const size_t need_fast = 40960ull + 2ull * NQ * BINS * DIMD * sizeof(__bf16);
  const int fast = (ws_size >= need_fast) ? 1 : 0;

  rvq_prep_kernel<<<dim3(NQ * BINS), dim3(256), 0, stream>>>(cb, norms, cbh, cbl, fast);
  if (fast) {
    rvq_main32_kernel<<<dim3((NB * TT) / ROWS2), dim3(256), 0, stream>>>(
        x, cb, norms, cbh, cbl, outq, outr, outc, partial);
    rvq_loss_kernel<<<dim3(1), dim3(256), 0, stream>>>(partial, (NB * TT) / ROWS2, outloss);
  } else {
    rvq_main16_kernel<<<dim3((NB * TT) / ROWS), dim3(256), 0, stream>>>(
        x, cb, norms, outq, outr, outc, partial);
    rvq_loss_kernel<<<dim3(1), dim3(256), 0, stream>>>(partial, (NB * TT) / ROWS, outloss);
  }
}
